// IDXCT_1133871366819
// MI455X (gfx1250) — matrix operand order repaired and checked by reading, not compiled, not run on hardware
//
#include <hip/hip_runtime.h>
#include <math.h>

// CDNA5 / gfx1250: wave32, WMMA 16x16x32 bf16 -> f32 accumulate.
typedef __attribute__((ext_vector_type(16))) __bf16 v16bf;
typedef __attribute__((ext_vector_type(8)))  __bf16 v8bf;
typedef __attribute__((ext_vector_type(8)))  float  v8f;
typedef __attribute__((ext_vector_type(4)))  float  f32x4;

#define TK    32        // K-step (one bf16 WMMA depth)
#define WGM   256       // workgroup output rows
#define WGN   128       // workgroup output cols (8 waves x 16 cols)
#define LDSB  40        // bf16 elements per padded LDS row (80 B, 16B-aligned)

// y[m,j] = sum_n x[m,n] * cos(pi * n * (2j+1) / (2N))
// GEMM where B is synthesized in-register: per-lane rotator table + phase state.
__global__ __launch_bounds__(256, 1)
void idxct_wmma_kernel(const float* __restrict__ x, float* __restrict__ y,
                       int M, int N)
{
    // A tile staged in bf16: 256 rows x 32 k, padded row stride 40 (80 B).
    __shared__ __align__(16) __bf16 As[WGM * LDSB];

    const int tid  = threadIdx.x;
    const int lane = tid & 31;
    const int wid  = tid >> 5;          // 0..7 : this wave's N-subtile
    const int half = lane >> 4;         // 0/1 lane group
    const int l16  = lane & 15;

    const int blockM = blockIdx.y * WGM;
    const int n0     = blockIdx.x * WGN + wid * 16;   // wave's 16-col strip
    const int col    = n0 + l16;                      // this lane's output column

    // Per-lane trig: alpha = pi*(2*col+1)/(2N).
    // Rotator table (loop-invariant): tc[e]=cos(e*alpha), ts[e]=sin(e*alpha).
    // Phase state (rc,rs) = (cos,sin)((kt + 16*half)*alpha), advanced by 32*alpha/step.
    float tc[16], ts[16];
    float rc, rs, c32, s32;
    {
        const double alpha = 3.14159265358979323846 * (double)(2 * col + 1)
                           / (2.0 * (double)N);
        #pragma unroll
        for (int e = 0; e < 16; ++e) {
            tc[e] = (float)cos(alpha * (double)e);
            ts[e] = (float)sin(alpha * (double)e);
        }
        c32 = (float)cos(alpha * 32.0);
        s32 = (float)sin(alpha * 32.0);
        const double th0 = alpha * (double)(16 * half);
        rc = (float)cos(th0);
        rs = (float)sin(th0);
    }

    v8f acc[16];  // 16 M-subtiles x (16x16 f32) = 128 VGPRs
    #pragma unroll
    for (int mt = 0; mt < 16; ++mt)
        acc[mt] = (v8f){0.f,0.f,0.f,0.f,0.f,0.f,0.f,0.f};

    for (int kt = 0; kt < N; kt += TK) {
        __syncthreads();
        // Stage + convert: 512 units of (1 row x 16 k); each thread does 2.
        #pragma unroll
        for (int i = 0; i < 2; ++i) {
            const int u   = tid + i * 256;
            const int row = u >> 1;
            const int h   = u & 1;
            const float* gp = x + (size_t)(blockM + row) * N + kt + h * 16;
            f32x4 v0 = ((const f32x4*)gp)[0];
            f32x4 v1 = ((const f32x4*)gp)[1];
            f32x4 v2 = ((const f32x4*)gp)[2];
            f32x4 v3 = ((const f32x4*)gp)[3];
            v8bf lo, hi;
            #pragma unroll
            for (int q = 0; q < 4; ++q) {
                lo[q]     = (__bf16)v0[q];
                lo[4 + q] = (__bf16)v1[q];
                hi[q]     = (__bf16)v2[q];
                hi[4 + q] = (__bf16)v3[q];
            }
            __bf16* dst = &As[row * LDSB + h * 16];
            *(v8bf*)(dst)     = lo;
            *(v8bf*)(dst + 8) = hi;
            if (kt + TK < N)
                __builtin_prefetch(gp + TK, 0, 1);   // global_prefetch next tile
        }
        __syncthreads();

        // B fragment (32x16 bf16): element e <-> k = kt + 16*half + e, col fixed.
        // b[e] = rc*tc[e] - rs*ts[e]  (independent mul+fmac pairs).
        v16bf bf;
        #pragma unroll
        for (int e = 0; e < 16; ++e) {
            const float ce = __builtin_fmaf(rc, tc[e], -(rs * ts[e]));
            bf[e] = (__bf16)ce;
        }
        {   // advance phase by 32*alpha for the next K-tile
            const float nrc = __builtin_fmaf(rc, c32, -(rs * s32));
            const float nrs = __builtin_fmaf(rs, c32,   rc * s32);
            rc = nrc; rs = nrs;
        }

        // 16 WMMAs share this B fragment; A frags are two aligned ds_load_b128.
        const int kb = half * 16;  // bf16 element offset within row
        #pragma unroll
        for (int mt = 0; mt < 16; ++mt) {
            const __bf16* rowp = &As[(mt * 16 + l16) * LDSB + kb];
            v8bf alo = *(const v8bf*)(rowp);        // e0..7  <-> k=kb..kb+7
            v8bf ahi = *(const v8bf*)(rowp + 8);    // e8..15 <-> k=kb+8..kb+15
            v16bf af = __builtin_shufflevector(alo, ahi,
                0,1,2,3,4,5,6,7, 8,9,10,11,12,13,14,15);
            acc[mt] = __builtin_amdgcn_wmma_f32_16x16x32_bf16(
                false, af, false, bf, (short)0, acc[mt], false, false);
        }
    }

    // Epilogue: C/D layout — VGPR r holds (M = r + 8*half, N = l16) per lane.
    #pragma unroll
    for (int mt = 0; mt < 16; ++mt) {
        #pragma unroll
        for (int r = 0; r < 8; ++r) {
            const int row = blockM + mt * 16 + half * 8 + r;
            y[(size_t)row * N + col] = acc[mt][r];
        }
    }
}

extern "C" void kernel_launch(void* const* d_in, const int* in_sizes, int n_in,
                              void* d_out, int out_size, void* d_ws, size_t ws_size,
                              hipStream_t stream)
{
    const float* x = (const float*)d_in[0];
    // in_sizes[1] = N*2 (expk [N,2]); expk is implied by N, recomputed on device.
    const int N = in_sizes[1] / 2;
    const int M = in_sizes[0] / N;
    float* y = (float*)d_out;

    dim3 grid(N / WGN, M / WGM);  // (128, 8) for 2048x16384
    dim3 block(256);
    idxct_wmma_kernel<<<grid, block, 0, stream>>>(x, y, M, N);
}
